// GIN_28956669510283
// MI455X (gfx1250) — matrix operand order repaired and checked by reading, not compiled, not run on hardware
//
#include <hip/hip_runtime.h>
#include <hip/hip_bf16.h>

#define HD 128
#define LLAYERS 3
#define WPITCH 136            // f16 pitch: 272B/row => bank offset 4/row, conflict-free b128 reads
#define BN_INV 0.9999950000374997f   // 1/sqrt(1+1e-5)

typedef __attribute__((ext_vector_type(16))) _Float16 v16h;
typedef __attribute__((ext_vector_type(8)))  _Float16 v8h;
typedef __attribute__((ext_vector_type(8)))  float    v8f;
typedef __attribute__((ext_vector_type(4)))  float    v4f;
typedef __attribute__((ext_vector_type(4)))  unsigned int u32x4;
typedef __attribute__((ext_vector_type(8)))  unsigned int u32x8;

// ---------------- embedding lookup: x[n][:] = z_table[z[n]][:] ----------------
__global__ void embed_kernel(const int* __restrict__ z, const float* __restrict__ table,
                             float* __restrict__ x, int n) {
    int t = blockIdx.x * blockDim.x + threadIdx.x;
    int node = t >> 5;
    int c = (t & 31) << 2;
    if (node >= n) return;
    v4f v = *(const v4f*)(table + (size_t)z[node] * HD + c);
    *(v4f*)(x + (size_t)node * HD + c) = v;
}

// ---------------- h = x (copy, before scatter-add) ----------------
__global__ void copy_kernel(const float* __restrict__ x, float* __restrict__ h, int n) {
    int t = blockIdx.x * blockDim.x + threadIdx.x;
    int node = t >> 5;
    int c = (t & 31) << 2;
    if (node >= n) return;
    *(v4f*)(h + (size_t)node * HD + c) = *(const v4f*)(x + (size_t)node * HD + c);
}

// ---------------- scatter-add: h[dst] += x[src]; one wave per edge ----------------
__global__ void scatter_kernel(const float* __restrict__ x, const int* __restrict__ src,
                               const int* __restrict__ dst, float* __restrict__ h, int ne) {
    int t = blockIdx.x * blockDim.x + threadIdx.x;
    int e = t >> 5;
    int c = (t & 31) << 2;
    if (e >= ne) return;
    int s = src[e], d = dst[e];
    v4f v = *(const v4f*)(x + (size_t)s * HD + c);
    float* hp = h + (size_t)d * HD + c;
    atomicAdd(hp + 0, v.x);
    atomicAdd(hp + 1, v.y);
    atomicAdd(hp + 2, v.z);
    atomicAdd(hp + 3, v.w);
}

// ---------------- weight pre-convert: f32 W[k][n] -> f16 Wt[n][k] (per matrix) ----------------
__global__ void wconv_kernel(const float* __restrict__ W, _Float16* __restrict__ Wt, int nmat) {
    int t = blockIdx.x * blockDim.x + threadIdx.x;
    if (t >= nmat * HD * HD) return;
    int m = t >> 14;               // 16384 elements per matrix
    int r = t & 16383;
    int k = r >> 7, n = r & (HD - 1);
    Wt[(size_t)m * HD * HD + (size_t)n * HD + k] = (_Float16)W[t];
}

// ---------------- fused GEMM + bias + ReLU (+ optional BN affine), WMMA f16->f32 ----------------
// Weights arrive pre-converted/transposed in f16 global (Wt_g[n][k], pitch 128) and are DMA'd
// into LDS by the Tensor Data Mover with pad_enable producing the 136-half conflict-free pitch.
__global__ __launch_bounds__(256)
void gemm_mlp_kernel(const float* __restrict__ in, const _Float16* __restrict__ Wt_g,
                     const float* __restrict__ bias, const float* __restrict__ gamma,
                     const float* __restrict__ beta, float* __restrict__ out,
                     float* __restrict__ out2, int n_rows) {
    __shared__ __align__(16) _Float16 Wt[HD * WPITCH];   // Wt[n][k] with pitch WPITCH

    if (threadIdx.x < 32) {
        // Tensor DMA descriptor (D#), 2-D tensor: 128 rows x 128 f16, pad 4 dwords / 64 dwords
        unsigned long long ga = (unsigned long long)(uintptr_t)Wt_g;
        unsigned lds = (unsigned)(uintptr_t)&Wt[0];      // LDS byte offset = addr[31:0]
        u32x4 g0;
        g0[0] = 1u;                                       // count=1, user descriptor
        g0[1] = lds;                                      // lds_addr
        g0[2] = (unsigned)(ga & 0xFFFFFFFFull);           // global_addr[31:0]
        g0[3] = (unsigned)((ga >> 32) & 0x01FFFFFFull)    // global_addr[56:32]
                | (2u << 30);                             // type = 2 ("image")
        u32x8 g1;
        g1[0] = (1u << 16)      // data_size = 1 -> 2 bytes
              | (1u << 20)      // pad_enable
              | (5u << 22)      // pad_interval = 5 -> every 64 dwords (256B row)
              | (3u << 25);     // pad_amount  = 3 -> 4 dwords (8 f16)  => pitch 136
        g1[1] = (128u << 16);   // tensor_dim0[15:0] = 128   (bits 63:48)
        g1[2] = (128u << 16);   // tensor_dim1[15:0] = 128   (bits 95:80)
        g1[3] = (128u << 16);   // tile_dim0 = 128           (bits 127:112)
        g1[4] = 128u;           // tile_dim1 = 128           (bits 143:128)
        g1[5] = 128u;           // tensor_dim0_stride = 128  (bits 191:160)
        g1[6] = 0u;
        g1[7] = 0u;
        asm volatile("tensor_load_to_lds %0, %1" :: "s"(g0), "s"(g1) : "memory");
        __builtin_amdgcn_s_wait_tensorcnt(0);
    }

    const int lane = threadIdx.x & 31;
    const int wave = threadIdx.x >> 5;
    const int half = lane >> 4;       // lane half selects K sub-groups per ISA layout
    const int l16  = lane & 15;
    const int rowBase = blockIdx.x * 128 + wave * 16;

    // ---- A fragments: 4 chunks of K=32, 16-bit A layout (16x32 MxK) ----
    v16h a[4];
    int arow = rowBase + l16;
    if (arow >= n_rows) arow = n_rows - 1;               // clamp; stores are guarded
    const float* ap = in + (size_t)arow * HD;
#pragma unroll
    for (int kc = 0; kc < 4; ++kc) {
        int k0 = kc * 32 + half * 8;
        v4f t4[4];
        t4[0] = *(const v4f*)(ap + k0);
        t4[1] = *(const v4f*)(ap + k0 + 4);
        t4[2] = *(const v4f*)(ap + k0 + 16);
        t4[3] = *(const v4f*)(ap + k0 + 20);
        const float* tmp = (const float*)t4;
#pragma unroll
        for (int e = 0; e < 16; ++e) a[kc][e] = (_Float16)tmp[e];
    }

    __syncthreads();   // Wt ready (TDM completed by wave 0 before its barrier arrival)

    const bool has_bn = (gamma != nullptr);
#pragma unroll
    for (int nt = 0; nt < 8; ++nt) {
        int ncol = nt * 16 + l16;
        float bv = bias[ncol];
        v8f c;
#pragma unroll
        for (int r = 0; r < 8; ++r) c[r] = bv;           // bias folded into accumulator
#pragma unroll
        for (int kc = 0; kc < 4; ++kc) {
            // B fragment (32x16 KxN): lane holds column N=l16, K = 8*half+[0..7] and 16+8*half+[0..7]
            const _Float16* bp = &Wt[(nt * 16 + l16) * WPITCH + kc * 32 + half * 8];
            v8h b0 = *(const v8h*)bp;
            v8h b1 = *(const v8h*)(bp + 16);
            v16h bf;
#pragma unroll
            for (int e = 0; e < 8; ++e) { bf[e] = b0[e]; bf[e + 8] = b1[e]; }
            c = __builtin_amdgcn_wmma_f32_16x16x32_f16(false, a[kc], false, bf,
                                                       (short)0, c, false, false);
        }
        float gm = 0.f, bt = 0.f;
        if (has_bn) { gm = gamma[ncol] * BN_INV; bt = beta[ncol]; }
#pragma unroll
        for (int r = 0; r < 8; ++r) {
            int row = rowBase + half * 8 + r;            // 32-bit C/D layout
            if (row < n_rows) {
                float v = c[r];
                v = v > 0.f ? v : 0.f;
                if (has_bn) v = gm * v + bt;
                out[(size_t)row * HD + ncol] = v;
                if (out2) out2[(size_t)row * (3 * HD) + ncol] = v;
            }
        }
    }
}

// ---------------- pooling helpers ----------------
__global__ void zero_kernel(float* __restrict__ p, int n) {
    int t = blockIdx.x * blockDim.x + threadIdx.x;
    if (t < n) p[t] = 0.f;
}

__global__ void count_kernel(const int* __restrict__ batch, float* __restrict__ cnt, int n) {
    int t = blockIdx.x * blockDim.x + threadIdx.x;
    if (t < n) atomicAdd(&cnt[batch[t]], 1.0f);
}

__global__ void poolsum_kernel(const float* __restrict__ xc, const int* __restrict__ batch,
                               float* __restrict__ sums, int n) {
    int t = blockIdx.x * blockDim.x + threadIdx.x;
    int node = t / 96;                // 384 channels / 4 per thread
    int j = (t - node * 96) * 4;
    if (node >= n) return;
    int g = batch[node];
    v4f v = *(const v4f*)(xc + (size_t)node * 384 + j);
    float* sp = sums + (size_t)g * 384 + j;
    atomicAdd(sp + 0, v.x);
    atomicAdd(sp + 1, v.y);
    atomicAdd(sp + 2, v.z);
    atomicAdd(sp + 3, v.w);
}

// ---------------- final MLP (tiny: 64x384 @ 384x128, then @128x1) ----------------
__global__ void mlp1_kernel(const float* __restrict__ sums, const float* __restrict__ cnt,
                            const float* __restrict__ Wm1, const float* __restrict__ bm1,
                            float* __restrict__ gbuf, int G) {
    int t = blockIdx.x * blockDim.x + threadIdx.x;
    if (t >= G * HD) return;
    int g = t >> 7, c = t & (HD - 1);
    float inv = 1.0f / fmaxf(cnt[g], 1.0f);
    const float* p = sums + (size_t)g * 3 * HD;
    float acc = bm1[c];
    for (int k = 0; k < 3 * HD; ++k) acc = fmaf(p[k] * inv, Wm1[(size_t)k * HD + c], acc);
    gbuf[t] = fmaxf(acc, 0.f);
}

__global__ void mlp2_kernel(const float* __restrict__ gbuf, const float* __restrict__ Wm2,
                            const float* __restrict__ bm2, float* __restrict__ out, int G) {
    int g = blockIdx.x * blockDim.x + threadIdx.x;
    if (g >= G) return;
    float acc = bm2[0];
    for (int c = 0; c < HD; ++c) acc = fmaf(gbuf[(size_t)g * HD + c], Wm2[c], acc);
    out[g] = acc;
}

extern "C" void kernel_launch(void* const* d_in, const int* in_sizes, int n_in,
                              void* d_out, int out_size, void* d_ws, size_t ws_size,
                              hipStream_t stream) {
    const int*   z       = (const int*)d_in[0];
    const int*   ei      = (const int*)d_in[1];     // [2,E]: src=ei[0:E], dst=ei[E:2E]
    const int*   batch   = (const int*)d_in[2];
    const float* z_table = (const float*)d_in[4];
    const float* W1s     = (const float*)d_in[5];
    const float* b1s     = (const float*)d_in[6];
    const float* W2s     = (const float*)d_in[7];
    const float* b2s     = (const float*)d_in[8];
    const float* gammas  = (const float*)d_in[9];
    const float* betas   = (const float*)d_in[10];
    const float* Wm1     = (const float*)d_in[11];
    const float* bm1     = (const float*)d_in[12];
    const float* Wm2     = (const float*)d_in[13];
    const float* bm2     = (const float*)d_in[14];

    const int N = in_sizes[0];
    const int E = in_sizes[1] / 2;
    const int G = out_size;

    // workspace layout (floats, then f16 weight images)
    float* ws   = (float*)d_ws;
    size_t NH   = (size_t)N * HD;
    float* x    = ws;                 // [N,128]
    float* h    = x + NH;             // [N,128]
    float* tbuf = h + NH;             // [N,128]
    float* xc   = tbuf + NH;          // [N,384] JK concat
    float* sums = xc + NH * 3;        // [G,384]
    float* cnt  = sums + (size_t)G * 3 * HD;  // [G]
    float* gbuf = cnt + G;            // [G,128]
    _Float16* wt1 = (_Float16*)(gbuf + (size_t)G * HD);  // [3][128][128] f16 (transposed)
    _Float16* wt2 = wt1 + (size_t)LLAYERS * HD * HD;     // [3][128][128] f16 (transposed)

    // one-time (per call) weight conversion to f16 transposed images for the TDM
    const int wthreads = LLAYERS * HD * HD;
    wconv_kernel<<<(wthreads + 255) / 256, 256, 0, stream>>>(W1s, wt1, LLAYERS);
    wconv_kernel<<<(wthreads + 255) / 256, 256, 0, stream>>>(W2s, wt2, LLAYERS);

    const int nvec = N * 32;                       // float4 threads over [N,128]
    embed_kernel<<<(nvec + 255) / 256, 256, 0, stream>>>(z, z_table, x, N);

    const int gemmBlocks = (N + 127) / 128;
    const int scatBlocks = (int)(((size_t)E * 32 + 255) / 256);
    for (int i = 0; i < LLAYERS; ++i) {
        copy_kernel<<<(nvec + 255) / 256, 256, 0, stream>>>(x, h, N);
        scatter_kernel<<<scatBlocks, 256, 0, stream>>>(x, ei, ei + E, h, E);
        // t = relu(h @ W1 + b1)
        gemm_mlp_kernel<<<gemmBlocks, 256, 0, stream>>>(
            h, wt1 + (size_t)i * HD * HD, b1s + i * HD, nullptr, nullptr, tbuf, nullptr, N);
        // x = gamma*BN_INV*relu(t @ W2 + b2) + beta ; also store into xc[:, i*H : (i+1)*H]
        gemm_mlp_kernel<<<gemmBlocks, 256, 0, stream>>>(
            tbuf, wt2 + (size_t)i * HD * HD, b2s + i * HD, gammas + i * HD, betas + i * HD,
            x, xc + i * HD, N);
    }

    const int zn = G * 3 * HD + G;   // sums + cnt are contiguous
    zero_kernel<<<(zn + 255) / 256, 256, 0, stream>>>(sums, zn);
    count_kernel<<<(N + 255) / 256, 256, 0, stream>>>(batch, cnt, N);
    poolsum_kernel<<<(N * 96 + 255) / 256, 256, 0, stream>>>(xc, batch, sums, N);
    mlp1_kernel<<<(G * HD + 255) / 256, 256, 0, stream>>>(sums, cnt, Wm1, bm1, gbuf, G);
    mlp2_kernel<<<(G + 63) / 64, 64, 0, stream>>>(gbuf, Wm2, bm2, (float*)d_out, G);
}
